// MoE_VulnerabilityDetector_24902220383016
// MI455X (gfx1250) — matrix operand order repaired and checked by reading, not compiled, not run on hardware
//
#include <hip/hip_runtime.h>
#include <hip/hip_bf16.h>
#include <math.h>

// ---------------------------------------------------------------------------
// MoE vulnerability detector, MI455X (gfx1250) wave32 + bf16 WMMA.
// Routed-sparse (top-2 of 8 experts), LN gains folded into bf16 weights,
// expert GEMMs on v_wmma_f32_16x16x32_bf16 with f32 accumulation.
// Each wave: 16x32 output tile (2 independent WMMA accumulator chains).
// ---------------------------------------------------------------------------

typedef __bf16 bf16_t;
typedef __bf16 v8bf  __attribute__((ext_vector_type(8)));
typedef __bf16 v16bf __attribute__((ext_vector_type(16)));
typedef float  v8f   __attribute__((ext_vector_type(8)));

#define B_   32768
#define D_   768
#define H_   256
#define H2_  128
#define E_   8
#define EPS_ 1e-5f

// ---------------- workspace layout (bytes) ----------------
static constexpr size_t WS_XNHAT = 0;                                   // B*D bf16
static constexpr size_t WS_WP1   = WS_XNHAT + (size_t)B_ * D_ * 2;      // 8*24*256*32 bf16
static constexpr size_t WS_B1EFF = WS_WP1   + (size_t)E_ * 24 * H_ * 32 * 2;
static constexpr size_t WS_WP2   = WS_B1EFF + (size_t)E_ * H_ * 4;      // 8*8*128*32 bf16
static constexpr size_t WS_B2EFF = WS_WP2   + (size_t)E_ * 8 * H2_ * 32 * 2;
static constexpr size_t WS_W3EFF = WS_B2EFF + (size_t)E_ * H2_ * 4;
static constexpr size_t WS_C3EFF = WS_W3EFF + (size_t)E_ * H2_ * 4;
static constexpr size_t WS_CNT   = WS_C3EFF + 256;
static constexpr size_t WS_PROB  = WS_CNT   + 64;
static constexpr size_t WS_OFFS  = WS_PROB  + 64;
static constexpr size_t WS_ROWS  = WS_OFFS  + 64;                       // E*B int
static constexpr size_t WS_WGT   = WS_ROWS  + (size_t)E_ * B_ * 4;      // E*B float
static constexpr size_t WS_H1    = WS_WGT   + (size_t)E_ * B_ * 4;      // 2B*H bf16
static constexpr size_t WS_H2    = WS_H1    + (size_t)2 * B_ * H_ * 2;  // 2B*H2 bf16

__device__ __forceinline__ float gelu_exact(float x) {
  return 0.5f * x * (1.0f + erff(x * 0.70710678118654752440f));
}

// ---------------- init: zero counters ----------------
__global__ void k_init(int* cnt, float* probAcc) {
  int t = threadIdx.x;
  if (t < E_) { cnt[t] = 0; probAcc[t] = 0.0f; }
}

// ---------------- weight prep: fold LN1 gain, pack B-fragment-major bf16 ----
// layout: Wp1[((e*24+kt)*256+col)*32 + j],  j = K offset within 32-wide ktile
__global__ __launch_bounds__(256) void k_prep_w1(const float* __restrict__ ln1_g,
                                                 const float* __restrict__ W1,
                                                 bf16_t* __restrict__ Wp1) {
  size_t idx = (size_t)blockIdx.x * 256 + threadIdx.x;   // 8*24*256*32
  int j = idx & 31; size_t t = idx >> 5;
  int col = t & 255; t >>= 8;
  int kt = (int)(t % 24); int e = (int)(t / 24);
  int k = kt * 32 + j;
  float v = ln1_g[e * D_ + k] * W1[((size_t)e * D_ + k) * H_ + col];
  Wp1[idx] = (bf16_t)v;
}

__global__ __launch_bounds__(256) void k_prep_b1(const float* __restrict__ ln1_b,
                                                 const float* __restrict__ W1,
                                                 const float* __restrict__ b1,
                                                 float* __restrict__ b1eff) {
  int idx = blockIdx.x * 256 + threadIdx.x;              // 8*256
  int h = idx & 255; int e = idx >> 8;
  float s = b1[e * H_ + h];
  for (int d = 0; d < D_; ++d)
    s += ln1_b[e * D_ + d] * W1[((size_t)e * D_ + d) * H_ + h];
  b1eff[idx] = s;
}

__global__ __launch_bounds__(256) void k_prep_w2(const float* __restrict__ ln2_g,
                                                 const float* __restrict__ W2,
                                                 bf16_t* __restrict__ Wp2) {
  size_t idx = (size_t)blockIdx.x * 256 + threadIdx.x;   // 8*8*128*32
  int j = idx & 31; size_t t = idx >> 5;
  int col = t & 127; t >>= 7;
  int kt = (int)(t & 7); int e = (int)(t >> 3);
  int k = kt * 32 + j;
  float v = ln2_g[e * H_ + k] * W2[((size_t)e * H_ + k) * H2_ + col];
  Wp2[idx] = (bf16_t)v;
}

__global__ __launch_bounds__(256) void k_prep_b2(const float* __restrict__ ln2_b,
                                                 const float* __restrict__ W2,
                                                 const float* __restrict__ b2,
                                                 float* __restrict__ b2eff) {
  int idx = blockIdx.x * 256 + threadIdx.x;              // 8*128
  int h = idx & 127; int e = idx >> 7;
  float s = b2[e * H2_ + h];
  for (int d = 0; d < H_; ++d)
    s += ln2_b[e * H_ + d] * W2[((size_t)e * H_ + d) * H2_ + h];
  b2eff[idx] = s;
}

__global__ __launch_bounds__(256) void k_prep_w3(const float* __restrict__ ln3_g,
                                                 const float* __restrict__ ln3_b,
                                                 const float* __restrict__ W3,
                                                 const float* __restrict__ b3,
                                                 float* __restrict__ w3eff,
                                                 float* __restrict__ c3eff) {
  int idx = blockIdx.x * 256 + threadIdx.x;              // 8*128
  w3eff[idx] = ln3_g[idx] * W3[idx];
  if (idx < E_) {
    float s = b3[idx];
    for (int h = 0; h < H2_; ++h) s += ln3_b[idx * H2_ + h] * W3[idx * H2_ + h];
    c3eff[idx] = s;
  }
}

// ---------------- fused input-LN + router-LN + logits + top-2 routing ------
__global__ __launch_bounds__(256) void k_ln_router(
    const float* __restrict__ x, const float* __restrict__ g_in,
    const float* __restrict__ b_in, const float* __restrict__ g_r,
    const float* __restrict__ b_r, const float* __restrict__ Wr,
    const float* __restrict__ br, bf16_t* __restrict__ xnhat,
    float* __restrict__ logitsOut, int* __restrict__ cnt,
    float* __restrict__ probAcc, int* __restrict__ rows,
    float* __restrict__ wgt, float* __restrict__ finalOut) {
  __shared__ float red[256];
  __shared__ float sh_mu, sh_rstd, sh_mu2, sh_rstd2;
  __shared__ float sh_logit[E_];
  int b = blockIdx.x, tid = threadIdx.x;
  const float* xr = x + (size_t)b * D_;

  float xv[3]; float s = 0.0f;
#pragma unroll
  for (int j = 0; j < 3; ++j) { xv[j] = xr[tid + 256 * j]; s += xv[j]; }
  red[tid] = s; __syncthreads();
  for (int st = 128; st; st >>= 1) { if (tid < st) red[tid] += red[tid + st]; __syncthreads(); }
  if (tid == 0) sh_mu = red[0] / D_;
  __syncthreads();
  float mu = sh_mu, q = 0.0f;
#pragma unroll
  for (int j = 0; j < 3; ++j) { float d = xv[j] - mu; q += d * d; }
  red[tid] = q; __syncthreads();
  for (int st = 128; st; st >>= 1) { if (tid < st) red[tid] += red[tid + st]; __syncthreads(); }
  if (tid == 0) sh_rstd = rsqrtf(red[0] / D_ + EPS_);
  __syncthreads();
  float rstd = sh_rstd;

  float xn[3]; float s2 = 0.0f;
#pragma unroll
  for (int j = 0; j < 3; ++j) {
    int d = tid + 256 * j;
    xn[j] = (xv[j] - mu) * rstd * g_in[d] + b_in[d];
    s2 += xn[j];
  }
  red[tid] = s2; __syncthreads();
  for (int st = 128; st; st >>= 1) { if (tid < st) red[tid] += red[tid + st]; __syncthreads(); }
  if (tid == 0) sh_mu2 = red[0] / D_;
  __syncthreads();
  float mu2 = sh_mu2, q2 = 0.0f;
#pragma unroll
  for (int j = 0; j < 3; ++j) { float d = xn[j] - mu2; q2 += d * d; }
  red[tid] = q2; __syncthreads();
  for (int st = 128; st; st >>= 1) { if (tid < st) red[tid] += red[tid + st]; __syncthreads(); }
  if (tid == 0) sh_rstd2 = rsqrtf(red[0] / D_ + EPS_);
  __syncthreads();
  float rstd2 = sh_rstd2;

  // normalized xn (shared by router LN and all expert LN1s) + router partials
  float part[E_];
#pragma unroll
  for (int e = 0; e < E_; ++e) part[e] = 0.0f;
#pragma unroll
  for (int j = 0; j < 3; ++j) {
    int d = tid + 256 * j;
    float hat = (xn[j] - mu2) * rstd2;
    xnhat[(size_t)b * D_ + d] = (bf16_t)hat;
    float rn = hat * g_r[d] + b_r[d];
#pragma unroll
    for (int e = 0; e < E_; ++e) part[e] += rn * Wr[e * D_ + d];
  }
  for (int e = 0; e < E_; ++e) {
    __syncthreads();
    red[tid] = part[e]; __syncthreads();
    for (int st = 128; st; st >>= 1) { if (tid < st) red[tid] += red[tid + st]; __syncthreads(); }
    if (tid == 0) sh_logit[e] = red[0] + br[e];
  }

  if (tid == 0) {
    float lg[E_], mx = -1e30f;
#pragma unroll
    for (int e = 0; e < E_; ++e) {
      lg[e] = sh_logit[e];
      logitsOut[(size_t)b * E_ + e] = lg[e];
      mx = fmaxf(mx, lg[e]);
    }
    float se = 0.0f, pe[E_];
#pragma unroll
    for (int e = 0; e < E_; ++e) { pe[e] = __expf(lg[e] - mx); se += pe[e]; }
#pragma unroll
    for (int e = 0; e < E_; ++e) atomicAdd(&probAcc[e], pe[e] / se);
    int i1 = 0;
    for (int e = 1; e < E_; ++e) if (lg[e] > lg[i1]) i1 = e;
    int i2 = (i1 == 0) ? 1 : 0;
    for (int e = 0; e < E_; ++e) if (e != i1 && lg[e] > lg[i2]) i2 = e;
    float e2 = __expf(lg[i2] - lg[i1]);
    float w1 = 1.0f / (1.0f + e2), w2 = e2 / (1.0f + e2);
    int s1 = atomicAdd(&cnt[i1], 1);
    rows[(size_t)i1 * B_ + s1] = b; wgt[(size_t)i1 * B_ + s1] = w1;
    int s2i = atomicAdd(&cnt[i2], 1);
    rows[(size_t)i2 * B_ + s2i] = b; wgt[(size_t)i2 * B_ + s2i] = w2;
    finalOut[b] = 0.0f;   // zero accumulator for layer-3 scatter-add
  }
}

// ---------------- finalize: aux outputs + prefix offsets ----------------
__global__ void k_finalize(const int* __restrict__ cnt,
                           const float* __restrict__ probAcc,
                           int* __restrict__ offs,
                           float* __restrict__ fracOut,
                           float* __restrict__ probOut) {
  int t = threadIdx.x;
  if (t < E_) {
    fracOut[t] = (float)cnt[t] / (float)B_;
    probOut[t] = probAcc[t] / (float)B_;
  }
  if (t == 0) {
    int o = 0;
    for (int e = 0; e < E_; ++e) { offs[e] = o; o += cnt[e]; }
    offs[E_] = o;
  }
}

// ---------------- grouped expert GEMM (bf16 WMMA, f32 acc) + GELU ---------
// One wave -> one 16x32 output tile: one A fragment feeds two independent
// WMMA accumulator chains (2 N-subtiles), doubling A reuse and giving the
// scheduler independent v_wmma issues. A: row-major bf16 (gathered rows for
// layer 1, compact slots for layer 2). B: fragment-major packed bf16 so each
// lane does one 32B contiguous load. A and B both use contiguous k = 16*half + i.
__global__ __launch_bounds__(256) void k_moe_gemm(
    const bf16_t* __restrict__ A, const bf16_t* __restrict__ Wp,
    const float* __restrict__ bias, bf16_t* __restrict__ Cout,
    const int* __restrict__ rowIdx,   // rows[e*B + s] gather, or nullptr
    const int* __restrict__ cnt, const int* __restrict__ offs,
    int lda, int N, int ktiles) {
  int e = blockIdx.z;
  int count = cnt[e];
  int wave = threadIdx.x >> 5, lane = threadIdx.x & 31;
  int mtile = blockIdx.x * 8 + wave;
  if (mtile * 16 >= count) return;
  int half = lane >> 4;            // K half selector (A and B)
  int l15 = lane & 15;
  int col = blockIdx.y * 32 + l15; // first of two 16-col subtiles

  int m_local = mtile * 16 + l15;
  int m_eff = m_local < count ? m_local : (count - 1);
  size_t rowA = rowIdx ? (size_t)rowIdx[(size_t)e * B_ + m_eff]
                       : (size_t)(offs[e] + m_eff);
  const bf16_t* Ap = A + rowA * lda + half * 16;
  const bf16_t* Bp = Wp + (size_t)e * ktiles * N * 32 + (size_t)col * 32 + half * 16;
  const size_t bStep = (size_t)N * 32;      // elements per ktile
  v8f acc0 = {}, acc1 = {};

  for (int kt = 0; kt < ktiles; ++kt) {
    // A 16x32 bf16: lane(m=l15): a[0..15]=K(half*16..+15)
    v8bf lo = *(const v8bf*)(Ap);
    v8bf hi = *(const v8bf*)(Ap + 8);
    v16bf a;
#pragma unroll
    for (int i = 0; i < 8; ++i) { a[i] = lo[i]; a[i + 8] = hi[i]; }
    // B 32x16 bf16: lane(col): b[i] = W[k0 + half*16 + i][col] -> contiguous 32B
    v16bf bf0 = *(const v16bf*)(Bp);
    v16bf bf1 = *(const v16bf*)(Bp + 16 * 32);
    acc0 = __builtin_amdgcn_wmma_f32_16x16x32_bf16(
        false, a, false, bf0, (short)0, acc0, false, false);
    acc1 = __builtin_amdgcn_wmma_f32_16x16x32_bf16(
        false, a, false, bf1, (short)0, acc1, false, false);
    Ap += 32;
    Bp += bStep;
  }

  float bn0 = bias[e * N + col];
  float bn1 = bias[e * N + col + 16];
  int slotBase = offs[e];
#pragma unroll
  for (int v = 0; v < 8; ++v) {
    int m = mtile * 16 + half * 8 + v;       // D layout: lanes>=16 hold M=8..15
    if (m < count) {
      size_t o = (size_t)(slotBase + m) * N + col;
      Cout[o]      = (bf16_t)gelu_exact(acc0[v] + bn0);
      Cout[o + 16] = (bf16_t)gelu_exact(acc1[v] + bn1);
    }
  }
}

// ---------------- in-place row LayerNorm-normalize (H=256, one wave/row) ---
__global__ __launch_bounds__(256) void k_rownorm256(bf16_t* __restrict__ h) {
  int wave = threadIdx.x >> 5, lane = threadIdx.x & 31;
  size_t row = (size_t)blockIdx.x * 8 + wave;
  bf16_t* p = h + row * H_;
  float vals[8]; float s = 0.0f;
#pragma unroll
  for (int j = 0; j < 8; ++j) { vals[j] = (float)p[lane + 32 * j]; s += vals[j]; }
  for (int m = 16; m; m >>= 1) s += __shfl_xor(s, m, 32);
  float mu = s / H_;
  float q = 0.0f;
#pragma unroll
  for (int j = 0; j < 8; ++j) { float d = vals[j] - mu; q += d * d; }
  for (int m = 16; m; m >>= 1) q += __shfl_xor(q, m, 32);
  float rstd = rsqrtf(q / H_ + EPS_);
#pragma unroll
  for (int j = 0; j < 8; ++j) p[lane + 32 * j] = (bf16_t)((vals[j] - mu) * rstd);
}

// ---------------- layer 3: LN3 + dot(w3eff) + weighted scatter-add --------
__global__ __launch_bounds__(256) void k_expert_out(
    const bf16_t* __restrict__ h2, const float* __restrict__ w3eff,
    const float* __restrict__ c3eff, const int* __restrict__ rows,
    const float* __restrict__ wgt, const int* __restrict__ cnt,
    const int* __restrict__ offs, float* __restrict__ finalOut) {
  int e = blockIdx.z;
  int count = cnt[e];
  int wave = threadIdx.x >> 5, lane = threadIdx.x & 31;
  int s = blockIdx.x * 8 + wave;
  if (s >= count) return;
  size_t slot = (size_t)offs[e] + s;
  const bf16_t* p = h2 + slot * H2_;
  float vals[4]; float sum = 0.0f;
#pragma unroll
  for (int j = 0; j < 4; ++j) { vals[j] = (float)p[lane + 32 * j]; sum += vals[j]; }
  for (int m = 16; m; m >>= 1) sum += __shfl_xor(sum, m, 32);
  float mu = sum / H2_;
  float q = 0.0f;
#pragma unroll
  for (int j = 0; j < 4; ++j) { float d = vals[j] - mu; q += d * d; }
  for (int m = 16; m; m >>= 1) q += __shfl_xor(q, m, 32);
  float rstd = rsqrtf(q / H2_ + EPS_);
  float acc = 0.0f;
#pragma unroll
  for (int j = 0; j < 4; ++j)
    acc += (vals[j] - mu) * rstd * w3eff[e * H2_ + lane + 32 * j];
  for (int m = 16; m; m >>= 1) acc += __shfl_xor(acc, m, 32);
  if (lane == 0) {
    float o = acc + c3eff[e];
    int b = rows[(size_t)e * B_ + s];
    atomicAdd(&finalOut[b], wgt[(size_t)e * B_ + s] * o);
  }
}

// ---------------------------------------------------------------------------
extern "C" void kernel_launch(void* const* d_in, const int* in_sizes, int n_in,
                              void* d_out, int out_size, void* d_ws, size_t ws_size,
                              hipStream_t stream) {
  const float* x     = (const float*)d_in[0];
  const float* g_in  = (const float*)d_in[1];
  const float* b_in  = (const float*)d_in[2];
  const float* g_r   = (const float*)d_in[3];
  const float* b_r   = (const float*)d_in[4];
  const float* Wr    = (const float*)d_in[5];
  const float* br    = (const float*)d_in[6];
  const float* ln1_g = (const float*)d_in[7];
  const float* ln1_b = (const float*)d_in[8];
  const float* W1    = (const float*)d_in[9];
  const float* b1    = (const float*)d_in[10];
  const float* ln2_g = (const float*)d_in[11];
  const float* ln2_b = (const float*)d_in[12];
  const float* W2    = (const float*)d_in[13];
  const float* b2    = (const float*)d_in[14];
  const float* ln3_g = (const float*)d_in[15];
  const float* ln3_b = (const float*)d_in[16];
  const float* W3    = (const float*)d_in[17];
  const float* b3    = (const float*)d_in[18];

  float* out       = (float*)d_out;
  float* out_final = out;                 // [B]
  float* out_frac  = out + B_;            // [E]
  float* out_prob  = out + B_ + E_;       // [E]
  float* out_logit = out + B_ + 2 * E_;   // [B*E]

  char* w = (char*)d_ws;
  bf16_t* xnhat  = (bf16_t*)(w + WS_XNHAT);
  bf16_t* Wp1    = (bf16_t*)(w + WS_WP1);
  float*  b1eff  = (float*)(w + WS_B1EFF);
  bf16_t* Wp2    = (bf16_t*)(w + WS_WP2);
  float*  b2eff  = (float*)(w + WS_B2EFF);
  float*  w3eff  = (float*)(w + WS_W3EFF);
  float*  c3eff  = (float*)(w + WS_C3EFF);
  int*    cnt    = (int*)(w + WS_CNT);
  float*  prob   = (float*)(w + WS_PROB);
  int*    offs   = (int*)(w + WS_OFFS);
  int*    rows   = (int*)(w + WS_ROWS);
  float*  wgt    = (float*)(w + WS_WGT);
  bf16_t* h1     = (bf16_t*)(w + WS_H1);
  bf16_t* h2     = (bf16_t*)(w + WS_H2);

  k_init<<<1, 32, 0, stream>>>(cnt, prob);

  k_prep_w1<<<(E_ * 24 * H_ * 32) / 256, 256, 0, stream>>>(ln1_g, W1, Wp1);
  k_prep_b1<<<(E_ * H_) / 256, 256, 0, stream>>>(ln1_b, W1, b1, b1eff);
  k_prep_w2<<<(E_ * 8 * H2_ * 32) / 256, 256, 0, stream>>>(ln2_g, W2, Wp2);
  k_prep_b2<<<(E_ * H2_) / 256, 256, 0, stream>>>(ln2_b, W2, b2, b2eff);
  k_prep_w3<<<(E_ * H2_) / 256, 256, 0, stream>>>(ln3_g, ln3_b, W3, b3, w3eff, c3eff);

  k_ln_router<<<B_, 256, 0, stream>>>(x, g_in, b_in, g_r, b_r, Wr, br, xnhat,
                                      out_logit, cnt, prob, rows, wgt, out_final);

  k_finalize<<<1, 32, 0, stream>>>(cnt, prob, offs, out_frac, out_prob);

  // Layer 1: [count_e x 768] @ [768 x 256], gathered rows, GELU, bf16 out
  k_moe_gemm<<<dim3(B_ / 16 / 8, H_ / 32, E_), 256, 0, stream>>>(
      xnhat, Wp1, b1eff, h1, rows, cnt, offs, D_, H_, D_ / 32);

  // LN2 normalize (gain folded into Wp2): 2B compact rows of 256
  k_rownorm256<<<(2 * B_) / 8, 256, 0, stream>>>(h1);

  // Layer 2: [count_e x 256] @ [256 x 128], compact slots, GELU, bf16 out
  k_moe_gemm<<<dim3(B_ / 16 / 8, H2_ / 32, E_), 256, 0, stream>>>(
      h1, Wp2, b2eff, h2, nullptr, cnt, offs, H_, H2_, H_ / 32);

  // Layer 3: LN3 + dot + weighted mix into final output
  k_expert_out<<<dim3(B_ / 8, 1, E_), 256, 0, stream>>>(
      h2, w3eff, c3eff, rows, wgt, cnt, offs, out_final);
}
